// MixtralSparseMoE_42949672960149
// MI455X (gfx1250) — compile-verified, matrix operand order checked by reading, not run on hardware
//
#include <hip/hip_runtime.h>
#include <math.h>

// ---------------------------------------------------------------------------
// Mixtral sparse MoE for MI455X (gfx1250), wave32, WMMA bf16 path.
//   T=2048 tokens, H=1024, I=3584, E=8, top-2 routing.
// Sparse (gathered) dispatch. 32-token tiles with 2x M register blocking:
// each w1/w3/w2 B-fragment feeds two v_wmma_f32_16x16x32_bf16 ops, halving
// weight bandwidth vs 16-token tiles. SwiGLU intermediate (32x3584 bf16
// ~224 KB) lives in the 320 KB WGP LDS so the expert MLP is one fused kernel.
// ---------------------------------------------------------------------------

typedef __bf16 bf16;
typedef __attribute__((ext_vector_type(16))) __bf16 v16bf;
typedef __attribute__((ext_vector_type(8)))  __bf16 v8bf;
typedef __attribute__((ext_vector_type(4)))  __bf16 v4bf;
typedef __attribute__((ext_vector_type(8)))  float  v8f;
typedef __attribute__((ext_vector_type(4)))  float  v4f;

constexpr int   H       = 1024;
constexpr int   I_DIM   = 3584;
constexpr int   E       = 8;
constexpr float EPSV    = 1e-6f;
constexpr int   TILE_M  = 32;            // tokens per expert tile (2 WMMA rows)
constexpr int   LDSPIT  = I_DIM + 8;     // bf16 row pitch (+8 dodges bank conflicts)

// ---------------------------------------------------------------------------
// Kernel 0: residual init (out = hidden) + zero expert counters.
// ---------------------------------------------------------------------------
__global__ void init_out_kernel(const float* __restrict__ hidden,
                                float* __restrict__ out,
                                int* __restrict__ counts, int n) {
  int i = blockIdx.x * blockDim.x + threadIdx.x;
  if (i < n) out[i] = hidden[i];
  if (i < E) counts[i] = 0;
}

// ---------------------------------------------------------------------------
// Kernel 1: RMSNorm (fp32) -> h_bf (bf16), router softmax/top2/renorm,
// expert gather-list construction. One block (256 thr) per token.
// ---------------------------------------------------------------------------
__global__ void __launch_bounds__(256)
rms_router_kernel(const float* __restrict__ hidden,
                  const float* __restrict__ rmsw,
                  const float* __restrict__ gate_w,
                  bf16*  __restrict__ h_bf,
                  float* __restrict__ cw,
                  int*   __restrict__ counts,
                  int*   __restrict__ token_list,
                  int T) {
  const int t   = blockIdx.x;
  const int tid = threadIdx.x;           // 256 threads, 4 elems each
  __shared__ float red[256];
  __shared__ float plog[256][E];
  __shared__ float s_scale;

  const float* xrow = hidden + (size_t)t * H;
  const int j0 = tid * 4;
  v4f xv = *(const v4f*)(xrow + j0);
  red[tid] = xv.x * xv.x + xv.y * xv.y + xv.z * xv.z + xv.w * xv.w;
  __syncthreads();
  for (int s = 128; s > 0; s >>= 1) {
    if (tid < s) red[tid] += red[tid + s];
    __syncthreads();
  }
  if (tid == 0) s_scale = rsqrtf(red[0] / (float)H + EPSV);
  __syncthreads();
  const float scale = s_scale;

  float h0 = rmsw[j0 + 0] * (xv.x * scale);
  float h1 = rmsw[j0 + 1] * (xv.y * scale);
  float h2 = rmsw[j0 + 2] * (xv.z * scale);
  float h3 = rmsw[j0 + 3] * (xv.w * scale);

  v4bf hv;
  hv[0] = (bf16)h0; hv[1] = (bf16)h1; hv[2] = (bf16)h2; hv[3] = (bf16)h3;
  *(v4bf*)(h_bf + (size_t)t * H + j0) = hv;

  #pragma unroll
  for (int e = 0; e < E; ++e) {
    const float* gw = gate_w + (size_t)e * H + j0;
    plog[tid][e] = h0 * gw[0] + h1 * gw[1] + h2 * gw[2] + h3 * gw[3];
  }
  __syncthreads();
  for (int s = 128; s > 0; s >>= 1) {
    if (tid < s) {
      #pragma unroll
      for (int e = 0; e < E; ++e) plog[tid][e] += plog[tid + s][e];
    }
    __syncthreads();
  }
  if (tid == 0) {
    float p[E];
    float mx = plog[0][0];
    #pragma unroll
    for (int e = 1; e < E; ++e) mx = fmaxf(mx, plog[0][e]);
    float se = 0.f;
    #pragma unroll
    for (int e = 0; e < E; ++e) { p[e] = __expf(plog[0][e] - mx); se += p[e]; }
    #pragma unroll
    for (int e = 0; e < E; ++e) p[e] /= se;
    // top-2 (ties -> lower index, matching lax.top_k)
    int i1 = 0;
    #pragma unroll
    for (int e = 1; e < E; ++e) if (p[e] > p[i1]) i1 = e;
    int i2 = (i1 == 0) ? 1 : 0;
    #pragma unroll
    for (int e = 0; e < E; ++e) if (e != i1 && p[e] > p[i2]) i2 = e;
    const float wsum = p[i1] + p[i2];
    const float wa = p[i1] / wsum, wb = p[i2] / wsum;
    #pragma unroll
    for (int e = 0; e < E; ++e)
      cw[(size_t)t * E + e] = (e == i1) ? wa : ((e == i2) ? wb : 0.f);
    int pos1 = atomicAdd(&counts[i1], 1);
    token_list[i1 * T + pos1] = t;
    int pos2 = atomicAdd(&counts[i2], 1);
    token_list[i2 * T + pos2] = t;
  }
}

// Helper: load two contiguous 8-float runs and pack to a v16bf B/A fragment.
__device__ __forceinline__ v16bf load_cvt_frag(const float* p, int k0) {
  v4f f0 = *(const v4f*)(p + k0);
  v4f f1 = *(const v4f*)(p + k0 + 4);
  v4f f2 = *(const v4f*)(p + k0 + 16);
  v4f f3 = *(const v4f*)(p + k0 + 20);
  v16bf r;
  #pragma unroll
  for (int q = 0; q < 4; ++q) {
    r[q]      = (bf16)f0[q];  r[4 + q]  = (bf16)f1[q];
    r[8 + q]  = (bf16)f2[q];  r[12 + q] = (bf16)f3[q];
  }
  return r;
}

__device__ __forceinline__ v16bf load_bf_frag(const bf16* p, int k0) {
  v8bf r0 = *(const v8bf*)(p + k0);
  v8bf r1 = *(const v8bf*)(p + k0 + 16);
  v16bf a;
  #pragma unroll
  for (int q = 0; q < 8; ++q) { a[q] = r0[q]; a[8 + q] = r1[q]; }
  return a;
}

// ---------------------------------------------------------------------------
// Kernel 2: fused sparse expert MLP. Block = (32-token tile, expert).
// Phase 1: act = silu(h @ w1^T) * (h @ w3^T)  -> LDS (bf16, padded pitch)
// Phase 2: out += cw * (act @ w2^T)           -> global atomic f32 add
// 256 threads = 8 wave32; each wave owns 16-wide N tiles round-robin and
// computes TWO 16x16 output tiles (token rows 0-15 and 16-31) per B load.
// ---------------------------------------------------------------------------
__global__ void __launch_bounds__(256)
expert_mlp_kernel(const bf16*  __restrict__ h_bf,
                  const float* __restrict__ w1,
                  const float* __restrict__ w2,
                  const float* __restrict__ w3,
                  const float* __restrict__ cw,
                  const int*   __restrict__ counts,
                  const int*   __restrict__ token_list,
                  float* __restrict__ out,
                  int T) {
  const int e    = blockIdx.y;
  const int tile = blockIdx.x;
  const int cnt  = counts[e];
  if (tile * TILE_M >= cnt) return;      // sparse: skip unrouted capacity

  __shared__ bf16  act[TILE_M][LDSPIT];  // ~224 KB of the 320 KB WGP LDS
  __shared__ int   s_tok[TILE_M];
  __shared__ float s_cw[TILE_M];

  const int tid  = threadIdx.x;
  const int wave = tid >> 5;             // 0..7
  const int lane = tid & 31;
  const int half = lane >> 4;            // K-half selector (WMMA layout)
  const int l16  = lane & 15;            // M (A/C rows) or N (B/C cols)

  if (tid < TILE_M) {
    int g = tile * TILE_M + tid;
    if (g < cnt) {
      int t = token_list[e * T + g];
      s_tok[tid] = t;
      s_cw[tid]  = cw[(size_t)t * E + e];
    } else {
      s_tok[tid] = 0;                    // harmless pad row
      s_cw[tid]  = 0.f;
    }
  }
  __syncthreads();

  const size_t wbase = (size_t)e * I_DIM * H;               // w1 / w3 expert base
  const bf16*  arow0 = h_bf + (size_t)s_tok[l16] * H;       // token rows 0-15
  const bf16*  arow1 = h_bf + (size_t)s_tok[l16 + 16] * H;  // token rows 16-31

  // ---------------- Phase 1: GEMM over H, fused SwiGLU ----------------
  for (int nt = wave; nt < I_DIM / 16; nt += 8) {
    const int ibase = nt * 16;
    const float* b1row = w1 + wbase + (size_t)(ibase + l16) * H;
    const float* b3row = w3 + wbase + (size_t)(ibase + l16) * H;
    v8f acc_a0 = {}, acc_a1 = {};
    v8f acc_g0 = {}, acc_g1 = {};
    for (int kb = 0; kb < H; kb += 32) {
      const int k0 = kb + half * 8;
      v16bf af0 = load_bf_frag(arow0, k0);
      v16bf af1 = load_bf_frag(arow1, k0);
      v16bf b1f = load_cvt_frag(b1row, k0);
      v16bf b3f = load_cvt_frag(b3row, k0);
      if (kb + 32 < H) {
        __builtin_prefetch(b1row + k0 + 32, 0, 3);   // near-cache prefetch
        __builtin_prefetch(b3row + k0 + 32, 0, 3);
      }
      // one B fragment feeds two M tiles: 4 WMMAs per (w1,w3) load pair
      acc_a0 = __builtin_amdgcn_wmma_f32_16x16x32_bf16(
          false, af0, false, b1f, (short)0, acc_a0, false, false);
      acc_a1 = __builtin_amdgcn_wmma_f32_16x16x32_bf16(
          false, af1, false, b1f, (short)0, acc_a1, false, false);
      acc_g0 = __builtin_amdgcn_wmma_f32_16x16x32_bf16(
          false, af0, false, b3f, (short)0, acc_g0, false, false);
      acc_g1 = __builtin_amdgcn_wmma_f32_16x16x32_bf16(
          false, af1, false, b3f, (short)0, acc_g1, false, false);
    }
    // fused silu(a) * g epilogue into LDS (C layout: row = v + 8*half, col = l16)
    #pragma unroll
    for (int v = 0; v < 8; ++v) {
      const int m = v + half * 8;
      const float a0 = acc_a0[v];
      const float s0 = (a0 / (1.f + __expf(-a0))) * acc_g0[v];
      act[m][ibase + l16] = (bf16)s0;
      const float a1 = acc_a1[v];
      const float s1 = (a1 / (1.f + __expf(-a1))) * acc_g1[v];
      act[m + 16][ibase + l16] = (bf16)s1;
    }
  }
  __syncthreads();

  // ---------------- Phase 2: GEMM over I against w2, combine ----------------
  for (int nt = wave; nt < H / 16; nt += 8) {
    const int hb = nt * 16;
    const float* b2row = w2 + (size_t)e * H * I_DIM + (size_t)(hb + l16) * I_DIM;
    v8f acc0 = {}, acc1 = {};
    for (int kb = 0; kb < I_DIM; kb += 32) {
      const int k0 = kb + half * 8;
      v16bf af0 = load_bf_frag(&act[l16][0], k0);        // rows 0-15 (LDS)
      v16bf af1 = load_bf_frag(&act[l16 + 16][0], k0);   // rows 16-31 (LDS)
      v16bf bfr = load_cvt_frag(b2row, k0);
      if (kb + 32 < I_DIM) __builtin_prefetch(b2row + k0 + 32, 0, 3);
      acc0 = __builtin_amdgcn_wmma_f32_16x16x32_bf16(
          false, af0, false, bfr, (short)0, acc0, false, false);
      acc1 = __builtin_amdgcn_wmma_f32_16x16x32_bf16(
          false, af1, false, bfr, (short)0, acc1, false, false);
    }
    #pragma unroll
    for (int v = 0; v < 8; ++v) {
      const int m = v + half * 8;
      const float c0 = s_cw[m];
      if (c0 != 0.f)  // pad rows carry cw==0
        atomicAdd(&out[(size_t)s_tok[m] * H + hb + l16], c0 * acc0[v]);
      const float c1 = s_cw[m + 16];
      if (c1 != 0.f)
        atomicAdd(&out[(size_t)s_tok[m + 16] * H + hb + l16], c1 * acc1[v]);
    }
  }
}

// ---------------------------------------------------------------------------
// Host-side launch.  d_in order: hidden, rms_weight, gate_w, w1, w2, w3.
// ws layout: h_bf (T*H bf16) | cw (T*E f32) | counts (pad 128B) | lists (E*T i32)
// ---------------------------------------------------------------------------
extern "C" void kernel_launch(void* const* d_in, const int* in_sizes, int n_in,
                              void* d_out, int out_size, void* d_ws, size_t ws_size,
                              hipStream_t stream) {
  const float* hidden = (const float*)d_in[0];
  const float* rmsw   = (const float*)d_in[1];
  const float* gate_w = (const float*)d_in[2];
  const float* w1     = (const float*)d_in[3];
  const float* w2     = (const float*)d_in[4];
  const float* w3     = (const float*)d_in[5];
  float* out = (float*)d_out;

  const int T = in_sizes[0] / H;   // 2048 tokens

  char* ws = (char*)d_ws;
  size_t off = 0;
  bf16* h_bf = (bf16*)(ws + off);   off += (size_t)T * H * sizeof(bf16);
  float* cw  = (float*)(ws + off);  off += (size_t)T * E * sizeof(float);
  int* counts = (int*)(ws + off);   off += 128;
  int* tlist  = (int*)(ws + off);   off += (size_t)E * T * sizeof(int);
  (void)ws_size; (void)n_in; (void)out_size;

  const int n = T * H;
  init_out_kernel<<<(n + 255) / 256, 256, 0, stream>>>(hidden, out, counts, n);
  rms_router_kernel<<<T, 256, 0, stream>>>(hidden, rmsw, gate_w, h_bf, cw,
                                           counts, tlist, T);
  dim3 grid((T + TILE_M - 1) / TILE_M, E);
  expert_mlp_kernel<<<grid, 256, 0, stream>>>(h_bf, w1, w2, w3, cw,
                                              counts, tlist, out, T);
}
